// SiteNetDIMAttentionBlock_69922067579683
// MI455X (gfx1250) — compile-verified, matrix operand order checked by reading, not run on hardware
//
#include <hip/hip_runtime.h>
#include <hip/hip_bf16.h>

typedef __attribute__((ext_vector_type(16))) _Float16 v16h;
typedef __attribute__((ext_vector_type(8)))  float    v8f;

#define NEGBIG   (-1e30f)
#define SENTINEL (-3.402823466e38f)

// ---------------------------------------------------------------------------
// Fragment load helper: CDNA5 16-bit WMMA A/B layout.
// lanes 0-15  : K = k0..k0+7    and k0+16..k0+23
// lanes 16-31 : K = k0+8..k0+15 and k0+24..k0+31
// rowptr points at the (M or N) row; kk = k0 + ((lane>>4)<<3).
// Emits two 16B loads (ds_load_b128 from LDS / global_load_b128 from global).
// ---------------------------------------------------------------------------
__device__ __forceinline__ v16h ldfrag(const _Float16* rowptr, int kk) {
  union { v16h v; uint4 q[2]; } u;
  u.q[0] = *(const uint4*)(rowptr + kk);
  u.q[1] = *(const uint4*)(rowptr + kk + 16);
  return u.v;
}

#define WMMA_F16(a, b, c) \
  __builtin_amdgcn_wmma_f32_16x16x32_f16(false, (a), false, (b), (short)0, (c), false, false)

// ---------------------------------------------------------------------------
// Linear(KD->256) + ReLU + LayerNorm on a 32-row tile, LN register-resident.
// A: LDS f16 [32][KD], Wt: global f16 [256][KD] (pre-transposed).
// 8 waves; wave w owns output columns [32w, 32w+32): two 16-col tiles for
// both 16-row tiles -> 4 WMMA accumulators; B fragments loaded once, used 2x.
// ---------------------------------------------------------------------------
template<int KD>
__device__ __forceinline__ void layer32(const _Float16* Ald, const _Float16* Wt,
                                        const float* bias, const float* gam, const float* bet,
                                        float* psum, float* pssq, float* mus, float* rss,
                                        _Float16* Hout, int wave, int lane)
{
  const int row   = lane & 15;
  const int khalf = (lane >> 4) << 3;
  const _Float16* arow0 = Ald + row * KD;              // rows 0..15
  const _Float16* arow1 = Ald + (16 + row) * KD;       // rows 16..31
  const _Float16* brow0 = Wt + (size_t)(wave * 32 + row) * KD;
  const _Float16* brow1 = brow0 + 16 * KD;

  v8f acc00 = {}, acc01 = {}, acc10 = {}, acc11 = {};
#pragma unroll
  for (int k0 = 0; k0 < KD; k0 += 32) {
    v16h a0 = ldfrag(arow0, k0 + khalf);
    v16h a1 = ldfrag(arow1, k0 + khalf);
    v16h b0 = ldfrag(brow0, k0 + khalf);
    v16h b1 = ldfrag(brow1, k0 + khalf);
    acc00 = WMMA_F16(a0, b0, acc00);
    acc01 = WMMA_F16(a0, b1, acc01);
    acc10 = WMMA_F16(a1, b0, acc10);
    acc11 = WMMA_F16(a1, b1, acc11);
  }

  // C/D layout: lanes 0-15 -> M=i, lanes 16-31 -> M=8+i ; N = lane&15 in tile.
  const int col0 = wave * 32 + row;
  const int col1 = col0 + 16;
  const float bb0 = bias[col0], bb1 = bias[col1];

  float z00[8], z01[8], z10[8], z11[8];
#pragma unroll
  for (int i = 0; i < 8; ++i) {
    z00[i] = fmaxf(acc00[i] + bb0, 0.f);
    z01[i] = fmaxf(acc01[i] + bb1, 0.f);
    z10[i] = fmaxf(acc10[i] + bb0, 0.f);
    z11[i] = fmaxf(acc11[i] + bb1, 0.f);
  }

  // Per-row partial sums over this wave's 32 columns: reduce across the
  // 16-lane half (xor 8,4,2,1 keeps bit4 fixed), one writer lane per row.
#pragma unroll
  for (int i = 0; i < 8; ++i) {
    float s0 = z00[i] + z01[i], q0 = z00[i] * z00[i] + z01[i] * z01[i];
    float s1 = z10[i] + z11[i], q1 = z10[i] * z10[i] + z11[i] * z11[i];
    for (int off = 8; off > 0; off >>= 1) {
      s0 += __shfl_xor(s0, off, 32); q0 += __shfl_xor(q0, off, 32);
      s1 += __shfl_xor(s1, off, 32); q1 += __shfl_xor(q1, off, 32);
    }
    if (row == i) {
      int r0 = khalf + i, r1 = 16 + khalf + i;
      psum[r0 * 8 + wave] = s0; pssq[r0 * 8 + wave] = q0;
      psum[r1 * 8 + wave] = s1; pssq[r1 * 8 + wave] = q1;
    }
  }
  __syncthreads();

  if (threadIdx.x < 32) {
    int r = threadIdx.x;
    float s = 0.f, q = 0.f;
    for (int wv = 0; wv < 8; ++wv) { s += psum[r * 8 + wv]; q += pssq[r * 8 + wv]; }
    float mu = s * (1.f / 256.f);
    mus[r] = mu;
    rss[r] = rsqrtf(q * (1.f / 256.f) - mu * mu + 1e-5f);
  }
  __syncthreads();

  const float gg0 = gam[col0], gg1 = gam[col1];
  const float be0 = bet[col0], be1 = bet[col1];
#pragma unroll
  for (int i = 0; i < 8; ++i) {
    int r0 = khalf + i, r1 = 16 + khalf + i;
    Hout[r0 * 256 + col0] = (_Float16)((z00[i] - mus[r0]) * rss[r0] * gg0 + be0);
    Hout[r0 * 256 + col1] = (_Float16)((z01[i] - mus[r0]) * rss[r0] * gg1 + be1);
    Hout[r1 * 256 + col0] = (_Float16)((z10[i] - mus[r1]) * rss[r1] * gg0 + be0);
    Hout[r1 * 256 + col1] = (_Float16)((z11[i] - mus[r1]) * rss[r1] * gg1 + be1);
  }
  __syncthreads();
}

// ---------------------------------------------------------------------------
// Weight transpose + fp32->f16 convert: in [K][Nc] -> out [Nc][K]
// ---------------------------------------------------------------------------
__global__ void k_convt(const float* __restrict__ in, _Float16* __restrict__ out, int K, int Nc) {
  int idx = blockIdx.x * 256 + threadIdx.x;
  if (idx >= K * Nc) return;
  int k = idx / Nc, n = idx % Nc;
  out[n * K + k] = (_Float16)in[idx];
}

// ---------------------------------------------------------------------------
// xs = LN(ReLU(x @ Wsite + bsite)) -> f16 [1024][64]. One block (64 thr)/row.
// ---------------------------------------------------------------------------
__global__ void __launch_bounds__(64) k_site(const float* __restrict__ x, const float* __restrict__ W,
                                             const float* __restrict__ b, const float* __restrict__ g,
                                             const float* __restrict__ t, _Float16* __restrict__ xs)
{
  __shared__ float xr[102];
  __shared__ float z[64];
  __shared__ float st[2];
  int n = blockIdx.x, tid = threadIdx.x;
  for (int i = tid; i < 102; i += 64) xr[i] = x[n * 102 + i];
  __syncthreads();
  float s = b[tid];
  for (int k = 0; k < 102; ++k) s += xr[k] * W[k * 64 + tid];
  s = fmaxf(s, 0.f);
  z[tid] = s;
  __syncthreads();
  if (tid == 0) {
    float m = 0.f, v = 0.f;
    for (int i = 0; i < 64; ++i) m += z[i];
    m *= (1.f / 64.f);
    for (int i = 0; i < 64; ++i) { float d = z[i] - m; v += d * d; }
    st[0] = m; st[1] = rsqrtf(v * (1.f / 64.f) + 1e-5f);
  }
  __syncthreads();
  xs[n * 64 + tid] = (_Float16)((s - st[0]) * st[1] * g[tid] + t[tid]);
}

// ---------------------------------------------------------------------------
// itf = LN(ReLU(if3 @ Wint + bint)) -> f16 [N*M][64]. One wave per row.
// ---------------------------------------------------------------------------
__global__ void __launch_bounds__(256) k_itf(const float* __restrict__ ifeat, const float* __restrict__ W,
                                             const float* __restrict__ b, const float* __restrict__ g,
                                             const float* __restrict__ t, _Float16* __restrict__ itf)
{
  int wave = threadIdx.x >> 5, lane = threadIdx.x & 31;
  size_t row = (size_t)blockIdx.x * 8 + wave;
  const float* f = ifeat + row * 3;
  float f0 = f[0], f1 = f[1], f2 = f[2];
  int c0 = lane, c1 = lane + 32;
  float z0 = fmaxf(b[c0] + f0 * W[c0] + f1 * W[64 + c0] + f2 * W[128 + c0], 0.f);
  float z1 = fmaxf(b[c1] + f0 * W[c1] + f1 * W[64 + c1] + f2 * W[128 + c1], 0.f);
  float s = z0 + z1, ss = z0 * z0 + z1 * z1;
  for (int off = 16; off > 0; off >>= 1) {
    s  += __shfl_xor(s,  off, 32);
    ss += __shfl_xor(ss, off, 32);
  }
  float mu = s * (1.f / 64.f);
  float rs = rsqrtf(ss * (1.f / 64.f) - mu * mu + 1e-5f);
  itf[row * 64 + c0] = (_Float16)((z0 - mu) * rs * g[c0] + t[c0]);
  itf[row * 64 + c1] = (_Float16)((z1 - mu) * rs * g[c1] + t[c1]);
}

// ---------------------------------------------------------------------------
// Main pairwise kernel: 32 pair-rows per block; both MLP branches via WMMA.
// grid = N*M/32 = 8192 blocks, 256 threads (8 waves).
// ---------------------------------------------------------------------------
__global__ void __launch_bounds__(256) k_pair(
    const _Float16* __restrict__ xs, const _Float16* __restrict__ itf,
    const int* __restrict__ ai, const int* __restrict__ aj,
    const int* __restrict__ amask, const float* __restrict__ ifeat,
    const _Float16* __restrict__ Wt1, const float* __restrict__ b1, const float* __restrict__ g1, const float* __restrict__ t1,
    const _Float16* __restrict__ Wt2, const float* __restrict__ b2, const float* __restrict__ g2, const float* __restrict__ t2,
    const float* __restrict__ Wp, const float* __restrict__ bp,
    const _Float16* __restrict__ At1, const float* __restrict__ ab1, const float* __restrict__ ag1, const float* __restrict__ at1,
    const _Float16* __restrict__ At2, const float* __restrict__ ab2, const float* __restrict__ ag2, const float* __restrict__ at2,
    const _Float16* __restrict__ At3, const float* __restrict__ ab3, const float* __restrict__ ag3, const float* __restrict__ at3,
    float* __restrict__ w_ws, _Float16* __restrict__ feats)
{
  __shared__ _Float16 xije[32 * 192];   // 12 KB
  __shared__ _Float16 h1[32 * 256];     // 16 KB
  __shared__ _Float16 h2[32 * 256];     // 16 KB
  __shared__ float    psum[32 * 8];     // 1 KB
  __shared__ float    pssq[32 * 8];     // 1 KB
  __shared__ float    mus[32], rss[32];
  __shared__ float    zb16[32 * 16];    // 2 KB (final feature layer)
  __shared__ float    mu16[32], rs16[32];

  const int tid  = threadIdx.x;
  const int wave = tid >> 5, lane = tid & 31;
  const int n  = blockIdx.x >> 3;
  const int m0 = (blockIdx.x & 7) << 5;

  // Build x_ije tile [32][192] f16 in LDS (gather xs[ai], xs[aj], itf).
  for (int idx = tid; idx < 32 * 96; idx += 256) {
    int r = idx / 96, w2 = idx % 96;            // 32-bit word within the row
    int pr = n * 256 + m0 + r;                  // global pair index
    unsigned int v;
    if (w2 < 32)       v = *(const unsigned int*)(xs + (size_t)ai[pr] * 64 + w2 * 2);
    else if (w2 < 64)  v = *(const unsigned int*)(xs + (size_t)aj[pr] * 64 + (w2 - 32) * 2);
    else               v = *(const unsigned int*)(itf + (size_t)pr * 64 + (w2 - 64) * 2);
    *(unsigned int*)(xije + r * 192 + w2 * 2) = v;
  }
  __syncthreads();

  // ---- scores branch (ije_to_multihead) ----
  layer32<192>(xije, Wt1, b1, g1, t1, psum, pssq, mus, rss, h1, wave, lane);
  layer32<256>(h1,   Wt2, b2, g2, t2, psum, pssq, mus, rss, h2, wave, lane);

  // projection to H=4 heads + mask
  if (tid < 128) {
    int r = tid >> 2, h = tid & 3;
    float s = bp[h];
    for (int k = 0; k < 256; ++k) s += (float)h2[r * 256 + k] * Wp[k * 4 + h];
    int pr = n * 256 + m0 + r;
    bool mk = (amask[pr] != 0) || (ifeat[(size_t)pr * 3] > 0.9f);
    w_ws[(size_t)(n * 4 + h) * 256 + (m0 + r)] = mk ? NEGBIG : s;
  }
  __syncthreads();

  // ---- features branch (ije_to_attention_features) ----
  layer32<192>(xije, At1, ab1, ag1, at1, psum, pssq, mus, rss, h1, wave, lane);
  layer32<256>(h1,   At2, ab2, ag2, at2, psum, pssq, mus, rss, h2, wave, lane);

  // final 256 -> 16 layer: one 16-col tile; waves 0/1 take the two row tiles.
  if (wave < 2) {
    const int row   = lane & 15;
    const int khalf = (lane >> 4) << 3;
    const _Float16* arow = h2  + (wave * 16 + row) * 256;
    const _Float16* brow = At3 + row * 256;
    v8f acc = {};
#pragma unroll
    for (int k0 = 0; k0 < 256; k0 += 32) {
      v16h a = ldfrag(arow, k0 + khalf);
      v16h b = ldfrag(brow, k0 + khalf);
      acc = WMMA_F16(a, b, acc);
    }
#pragma unroll
    for (int i = 0; i < 8; ++i)
      zb16[(wave * 16 + khalf + i) * 16 + row] = fmaxf(acc[i] + ab3[row], 0.f);
  }
  __syncthreads();

  if (tid < 32) {
    float s = 0.f, ss = 0.f;
    for (int c = 0; c < 16; ++c) { float z = zb16[tid * 16 + c]; s += z; ss += z * z; }
    float mu = s * (1.f / 16.f);
    mu16[tid] = mu;
    rs16[tid] = rsqrtf(ss * (1.f / 16.f) - mu * mu + 1e-5f);
  }
  __syncthreads();
#pragma unroll
  for (int idx = tid; idx < 512; idx += 256) {
    int r = idx >> 4, c = idx & 15;
    float z = zb16[r * 16 + c];
    feats[(size_t)(n * 256 + m0 + r) * 16 + c] = (_Float16)((z - mu16[r]) * rs16[r] * ag3[c] + at3[c]);
  }
}

// ---------------------------------------------------------------------------
// top-K(32) + softmax + attn x feats einsum + final 64x64 linear.
// One block per site n; one wave per head.
// ---------------------------------------------------------------------------
__global__ void __launch_bounds__(128) k_attn(const float* __restrict__ w_ws, const _Float16* __restrict__ feats,
                                              const float* __restrict__ Wg, const float* __restrict__ bg,
                                              float* __restrict__ out)
{
  __shared__ float attn_s[4 * 256];
  __shared__ float ovec[64];
  const int n = blockIdx.x;
  const int tid = threadIdx.x, h = tid >> 5, lane = tid & 31;
  const float* wrow = w_ws + (size_t)(n * 4 + h) * 256;

  float o[8], v[8];
#pragma unroll
  for (int j = 0; j < 8; ++j) { o[j] = wrow[lane + 32 * j]; v[j] = o[j]; }

  float kth = NEGBIG, gmax = NEGBIG;
  for (int it = 0; it < 32; ++it) {
    float lm = SENTINEL; int li = 0x7fffffff;
#pragma unroll
    for (int j = 0; j < 8; ++j)
      if (v[j] > lm) { lm = v[j]; li = lane + 32 * j; }
    for (int off = 16; off > 0; off >>= 1) {
      float ol = __shfl_xor(lm, off, 32);
      int   oi = __shfl_xor(li, off, 32);
      if (ol > lm || (ol == lm && oi < li)) { lm = ol; li = oi; }
    }
    if (it == 0) gmax = lm;
    kth = lm;
    if (((li & 31) == lane) && li < 256) v[li >> 5] = SENTINEL;
  }

  float e[8], s = 0.f;
#pragma unroll
  for (int j = 0; j < 8; ++j) {
    float x = (o[j] >= kth) ? __expf(o[j] - gmax) : 0.f;
    e[j] = x; s += x;
  }
  for (int off = 16; off > 0; off >>= 1) s += __shfl_xor(s, off, 32);
  float inv = 1.f / s;
#pragma unroll
  for (int j = 0; j < 8; ++j) attn_s[h * 256 + lane + 32 * j] = e[j] * inv;
  __syncthreads();

  if (tid < 64) {
    int ee = tid >> 2, hh = tid & 3;
    const float* arow = attn_s + hh * 256;
    float acc = 0.f;
    for (int m = 0; m < 256; ++m)
      acc += arow[m] * (float)feats[(size_t)(n * 256 + m) * 16 + ee];
    ovec[ee * 4 + hh] = acc;   // reshape [DH,H] -> flat
  }
  __syncthreads();
  if (tid < 64) {
    float acc = bg[tid];
    for (int k = 0; k < 64; ++k) acc += ovec[k] * Wg[k * 64 + tid];
    out[(size_t)n * 64 + tid] = acc;
  }
}

// ---------------------------------------------------------------------------
extern "C" void kernel_launch(void* const* d_in, const int* in_sizes, int n_in,
                              void* d_out, int out_size, void* d_ws, size_t ws_size,
                              hipStream_t stream) {
  (void)in_sizes; (void)n_in; (void)out_size; (void)ws_size;
  const int N = 1024, M = 256;

  // inputs (setup_inputs dict order, params flattened in insertion order)
  const float* x       = (const float*)d_in[0];
  const float* ifeat   = (const float*)d_in[1];
  const int*   amask   = (const int*)d_in[2];
  const int*   ai      = (const int*)d_in[3];
  const int*   aj      = (const int*)d_in[4];
  const float* Wsite   = (const float*)d_in[5];
  const float* bsite   = (const float*)d_in[6];
  const float* gsite   = (const float*)d_in[7];
  const float* tsite   = (const float*)d_in[8];
  const float* Wint    = (const float*)d_in[9];
  const float* bint    = (const float*)d_in[10];
  const float* gint    = (const float*)d_in[11];
  const float* tint    = (const float*)d_in[12];
  const float* W1      = (const float*)d_in[13];
  const float* b1      = (const float*)d_in[14];
  const float* g1      = (const float*)d_in[15];
  const float* t1      = (const float*)d_in[16];
  const float* W2      = (const float*)d_in[17];
  const float* b2      = (const float*)d_in[18];
  const float* g2      = (const float*)d_in[19];
  const float* t2      = (const float*)d_in[20];
  const float* Wp      = (const float*)d_in[21];
  const float* bp      = (const float*)d_in[22];
  const float* A1      = (const float*)d_in[23];
  const float* ab1     = (const float*)d_in[24];
  const float* ag1     = (const float*)d_in[25];
  const float* at1     = (const float*)d_in[26];
  const float* A2      = (const float*)d_in[27];
  const float* ab2     = (const float*)d_in[28];
  const float* ag2     = (const float*)d_in[29];
  const float* at2     = (const float*)d_in[30];
  const float* A3      = (const float*)d_in[31];
  const float* ab3     = (const float*)d_in[32];
  const float* ag3     = (const float*)d_in[33];
  const float* at3     = (const float*)d_in[34];
  const float* Wg      = (const float*)d_in[35];
  const float* bg      = (const float*)d_in[36];
  float* out = (float*)d_out;

  // workspace carve-out (256B-aligned)
  char* ws = (char*)d_ws;
  size_t off = 0;
  auto carve = [&](size_t bytes) -> void* {
    off = (off + 255) & ~(size_t)255;
    void* p = ws + off;
    off += bytes;
    return p;
  };
  _Float16* Wt1 = (_Float16*)carve((size_t)256 * 192 * 2);
  _Float16* Wt2 = (_Float16*)carve((size_t)256 * 256 * 2);
  _Float16* At1 = (_Float16*)carve((size_t)256 * 192 * 2);
  _Float16* At2 = (_Float16*)carve((size_t)256 * 256 * 2);
  _Float16* At3 = (_Float16*)carve((size_t)16  * 256 * 2);
  _Float16* xsF = (_Float16*)carve((size_t)N * 64 * 2);
  _Float16* itf = (_Float16*)carve((size_t)N * M * 64 * 2);
  float*    wws = (float*)   carve((size_t)N * 4 * M * 4);
  _Float16* fts = (_Float16*)carve((size_t)N * M * 16 * 2);

  // 1) weight transpose/convert
  k_convt<<<(192 * 256 + 255) / 256, 256, 0, stream>>>(W1, Wt1, 192, 256);
  k_convt<<<(256 * 256 + 255) / 256, 256, 0, stream>>>(W2, Wt2, 256, 256);
  k_convt<<<(192 * 256 + 255) / 256, 256, 0, stream>>>(A1, At1, 192, 256);
  k_convt<<<(256 * 256 + 255) / 256, 256, 0, stream>>>(A2, At2, 256, 256);
  k_convt<<<(256 * 16  + 255) / 256, 256, 0, stream>>>(A3, At3, 256, 16);

  // 2) site embedding + interaction embedding
  k_site<<<N, 64, 0, stream>>>(x, Wsite, bsite, gsite, tsite, xsF);
  k_itf<<<N * M / 8, 256, 0, stream>>>(ifeat, Wint, bint, gint, tint, itf);

  // 3) pairwise MLPs (WMMA, 32-row tiles)
  k_pair<<<N * M / 32, 256, 0, stream>>>(xsF, itf, ai, aj, amask, ifeat,
                                         Wt1, b1, g1, t1, Wt2, b2, g2, t2, Wp, bp,
                                         At1, ab1, ag1, at1, At2, ab2, ag2, at2,
                                         At3, ab3, ag3, at3, wws, fts);

  // 4) top-K softmax + einsum + final linear
  k_attn<<<N, 128, 0, stream>>>(wws, fts, Wg, bg, out);
}
